// DecoderBlock_17102559773269
// MI455X (gfx1250) — matrix operand order repaired and checked by reading, not compiled, not run on hardware
//
#include <hip/hip_runtime.h>
#include <cstdint>

// ---------------------------------------------------------------------------
// GPT-2 style decoder block for MI455X (gfx1250, wave32, WMMA).
// B=4, S=1024, D=1024, H=16, hd=64.  All matmuls run through
// v_wmma_f32_16x16x32_f16 with on-the-fly f32->f16 conversion in LDS.
// GEMM uses double-buffered LDS + register software pipelining + prefetch.
// Workspace requirement: 40M floats = 160 MB.
// ---------------------------------------------------------------------------

typedef __attribute__((ext_vector_type(16))) _Float16 v16h;
typedef __attribute__((ext_vector_type(8)))  float    v8f;

union FragH {
  v16h    h;
  uint4   u4[2];
  _Float16 e[16];
};

union Pack4 {
  _Float16 h[4];
  uint2    u;
};

__device__ __forceinline__ float gelu_tanh(float x) {
  float x3 = x * x * x;
  float t  = tanhf(0.7978845608028654f * (x + 0.044715f * x3));
  return 0.5f * x * (1.0f + t);
}

// ---------------------------------------------------------------------------
// GEMM: C[M,N] = act( A[M,K] @ W[K,N] + bias[N] )   (all f32 in memory,
// f16 WMMA compute, f32 accumulate).  M,N multiples of 128; K multiple of 32.
// Double-buffered LDS, one barrier per K-step.
// ---------------------------------------------------------------------------
#define BM 128
#define BN 128
#define BK 32

__global__ __launch_bounds__(256) void gemm_f16_kernel(
    const float* __restrict__ A, const float* __restrict__ W,
    const float* __restrict__ bias, float* __restrict__ C,
    int M, int N, int K, int act_gelu)
{
  __shared__ _Float16 As[2][BM * BK];   // [m][k]
  __shared__ _Float16 Wt[2][BN * BK];   // [n][k] (transposed tile)

  const int tid  = threadIdx.x;
  const int lane = tid & 31;
  const int wave = tid >> 5;
  const int wr   = wave >> 1;   // 0..3 : 32-row group
  const int wc   = wave & 1;    // 0..1 : 64-col group
  const int ml   = lane & 15;
  const int lh   = lane >> 4;

  const int tileM = blockIdx.y * BM;
  const int tileN = blockIdx.x * BN;

  const v8f vzero = {0.f,0.f,0.f,0.f,0.f,0.f,0.f,0.f};
  v8f acc[2][4];
#pragma unroll
  for (int i = 0; i < 2; ++i)
#pragma unroll
    for (int j = 0; j < 4; ++j) acc[i][j] = vzero;

  const int ar  = tid >> 3;   // 0..31 : A row subgroup
  const int akq = tid & 7;    // float4 slot within A row
  const int wkr = tid >> 5;   // 0..7  : W k-row subgroup
  const int wnq = tid & 31;   // float4 slot within W row

  const float* Abase = A + (size_t)tileM * K;
  const float* Wbase = W + tileN;

  float4 aReg[4], wReg[4];

  auto load_tile = [&](int k0) {
#pragma unroll
    for (int i = 0; i < 4; ++i)
      aReg[i] = *(const float4*)&Abase[(size_t)(ar + i * 32) * K + k0 + akq * 4];
#pragma unroll
    for (int i = 0; i < 4; ++i)
      wReg[i] = *(const float4*)&Wbase[(size_t)(k0 + wkr + i * 8) * N + wnq * 4];
  };

  auto store_tile = [&](int buf) {
#pragma unroll
    for (int i = 0; i < 4; ++i) {
      Pack4 p;
      p.h[0] = (_Float16)aReg[i].x; p.h[1] = (_Float16)aReg[i].y;
      p.h[2] = (_Float16)aReg[i].z; p.h[3] = (_Float16)aReg[i].w;
      *(uint2*)&As[buf][(ar + i * 32) * BK + akq * 4] = p.u;
    }
#pragma unroll
    for (int i = 0; i < 4; ++i) {
      int k = wkr + i * 8;
      Wt[buf][(wnq * 4 + 0) * BK + k] = (_Float16)wReg[i].x;
      Wt[buf][(wnq * 4 + 1) * BK + k] = (_Float16)wReg[i].y;
      Wt[buf][(wnq * 4 + 2) * BK + k] = (_Float16)wReg[i].z;
      Wt[buf][(wnq * 4 + 3) * BK + k] = (_Float16)wReg[i].w;
    }
  };

  // prologue: stage tile 0
  load_tile(0);
  store_tile(0);

  int cur = 0;
  for (int k0 = 0; k0 < K; k0 += BK) {
    __syncthreads();   // single barrier per K-step (double buffer)

    const bool more = (k0 + BK) < K;
    if (more) load_tile(k0 + BK);           // global loads overlap WMMAs below
    if (k0 + 2 * BK < K) {                  // pull tile t+2 into GL2 early
      __builtin_prefetch(&Abase[(size_t)ar * K + k0 + 2 * BK], 0, 1);
      __builtin_prefetch(&Wbase[(size_t)(k0 + 2 * BK + wkr) * N + wnq * 4], 0, 1);
    }

    // A fragments (16x32, documented CDNA5 A layout)
    FragH af[2];
#pragma unroll
    for (int i = 0; i < 2; ++i) {
      int m  = wr * 32 + i * 16 + ml;
      int ko = lh * 8;
      af[i].u4[0] = *(const uint4*)&As[cur][m * BK + ko];        // K ko..ko+7
      af[i].u4[1] = *(const uint4*)&As[cur][m * BK + ko + 16];   // K ko+16..+23
    }
  // B fragments (32x16, lanes 0-15: K0-7,16-23, lanes 16-31: K8-15,24-31)
    FragH bf[4];
#pragma unroll
    for (int j = 0; j < 4; ++j) {
      int n  = wc * 64 + j * 16 + ml;
      int kb = lh * 8;
      bf[j].u4[0] = *(const uint4*)&Wt[cur][n * BK + kb];
      bf[j].u4[1] = *(const uint4*)&Wt[cur][n * BK + kb + 16];
    }
#pragma unroll
    for (int i = 0; i < 2; ++i)
#pragma unroll
      for (int j = 0; j < 4; ++j)
        acc[i][j] = __builtin_amdgcn_wmma_f32_16x16x32_f16(
            false, af[i].h, false, bf[j].h, (short)0, acc[i][j], false, false);

    if (more) store_tile(cur ^ 1);          // fill the other buffer
    cur ^= 1;
  }

  // ---- epilogue: bias + optional GELU ----
#pragma unroll
  for (int i = 0; i < 2; ++i) {
    const int rbase = tileM + wr * 32 + i * 16 + lh * 8;
#pragma unroll
    for (int j = 0; j < 4; ++j) {
      const int col = tileN + wc * 64 + j * 16 + ml;
      const float bv = bias[col];
#pragma unroll
      for (int r = 0; r < 8; ++r) {
        float v = acc[i][j][r] + bv;
        if (act_gelu) v = gelu_tanh(v);
        C[(size_t)(rbase + r) * N + col] = v;
      }
    }
  }
}

// ---------------------------------------------------------------------------
// Flash attention: one WG = 64 query rows of one (b,h); 4 waves x 16 rows.
// K processed in chunks of 32 keys. hd fixed = 64.
// ---------------------------------------------------------------------------
__global__ __launch_bounds__(128) void flash_attn_kernel(
    const float* __restrict__ Qb, const float* __restrict__ Kb,
    const float* __restrict__ Vb, float* __restrict__ Ob,
    int S, int strideQ, int strideKV, int strideO, int H, int causal)
{
  __shared__ _Float16 Ks[32 * 64];     // [key][d]
  __shared__ _Float16 Vt[64 * 32];     // [d][key]   (transposed)
  __shared__ _Float16 Ps[4][16 * 32];  // per-wave P relayout bounce

  const int tid  = threadIdx.x;
  const int lane = tid & 31;
  const int wave = tid >> 5;
  const int ml   = lane & 15;
  const int lh   = lane >> 4;

  const int bh   = blockIdx.y;
  const int b    = bh / H;
  const int h    = bh % H;
  const int qblk = blockIdx.x * 64;
  const int q0   = qblk + wave * 16;

  const float scale = 0.125f;  // 1/sqrt(64)

  const float* Q  = Qb + (size_t)b * S * strideQ  + h * 64;
  const float* Kp = Kb + (size_t)b * S * strideKV + h * 64;
  const float* Vp = Vb + (size_t)b * S * strideKV + h * 64;
  float*       O  = Ob + (size_t)b * S * strideO  + h * 64;

  // ---- Q fragments (16x64 -> two 16x32 A-frags), scale folded in ----
  FragH qf[2];
  {
    const int qrow = q0 + ml;
    const int ko   = lh * 8;
#pragma unroll
    for (int kt = 0; kt < 2; ++kt) {
      const float* qp = &Q[(size_t)qrow * strideQ + kt * 32 + ko];
      float4 f0 = *(const float4*)(qp);
      float4 f1 = *(const float4*)(qp + 4);
      float4 f2 = *(const float4*)(qp + 16);
      float4 f3 = *(const float4*)(qp + 20);
      float tv[16] = {f0.x,f0.y,f0.z,f0.w, f1.x,f1.y,f1.z,f1.w,
                      f2.x,f2.y,f2.z,f2.w, f3.x,f3.y,f3.z,f3.w};
#pragma unroll
      for (int e = 0; e < 16; ++e) qf[kt].e[e] = (_Float16)(tv[e] * scale);
    }
  }

  const v8f vzero = {0.f,0.f,0.f,0.f,0.f,0.f,0.f,0.f};
  v8f o[4];
#pragma unroll
  for (int dj = 0; dj < 4; ++dj) o[dj] = vzero;
  float mrow[8], lrow[8];
#pragma unroll
  for (int r = 0; r < 8; ++r) { mrow[r] = -3.0e38f; lrow[r] = 0.f; }

  const int jend = causal ? (qblk + 64) : S;
  for (int j0 = 0; j0 < jend; j0 += 32) {
    __syncthreads();
    // ---- stage K (row-major) and V (transposed) chunk, f32 -> f16 ----
#pragma unroll
    for (int i = 0; i < 4; ++i) {
      int idx = tid + i * 128;          // 0..511 float4 units
      int key = idx >> 4;
      int dq  = idx & 15;
      const float4 kf = *(const float4*)&Kp[(size_t)(j0 + key) * strideKV + dq * 4];
      Pack4 pk;
      pk.h[0]=(_Float16)kf.x; pk.h[1]=(_Float16)kf.y;
      pk.h[2]=(_Float16)kf.z; pk.h[3]=(_Float16)kf.w;
      *(uint2*)&Ks[key * 64 + dq * 4] = pk.u;
      const float4 vf = *(const float4*)&Vp[(size_t)(j0 + key) * strideKV + dq * 4];
      Vt[(dq * 4 + 0) * 32 + key] = (_Float16)vf.x;
      Vt[(dq * 4 + 1) * 32 + key] = (_Float16)vf.y;
      Vt[(dq * 4 + 2) * 32 + key] = (_Float16)vf.z;
      Vt[(dq * 4 + 3) * 32 + key] = (_Float16)vf.w;
    }
    __syncthreads();

    // ---- scores: S = Q @ K^T  (two 16x16 tiles, K-dim 64 = 2 steps) ----
    v8f sc[2];
    sc[0] = vzero; sc[1] = vzero;
#pragma unroll
    for (int j = 0; j < 2; ++j) {
#pragma unroll
      for (int kt = 0; kt < 2; ++kt) {
        FragH bk;
        int key = j * 16 + ml;
        int dof = kt * 32 + lh * 8;
        bk.u4[0] = *(const uint4*)&Ks[key * 64 + dof];
        bk.u4[1] = *(const uint4*)&Ks[key * 64 + dof + 16];
        sc[j] = __builtin_amdgcn_wmma_f32_16x16x32_f16(
            false, qf[kt].h, false, bk.h, (short)0, sc[j], false, false);
      }
    }

    // ---- causal mask (matches reference: masked -> -10000) ----
    if (causal && (j0 + 31 > q0)) {
#pragma unroll
      for (int j = 0; j < 2; ++j) {
        int col = j0 + j * 16 + ml;
#pragma unroll
        for (int r = 0; r < 8; ++r) {
          int row = q0 + lh * 8 + r;
          if (col > row) sc[j][r] = -10000.0f;
        }
      }
    }

    // ---- online softmax update (rows live in VGPR idx + lane half) ----
#pragma unroll
    for (int r = 0; r < 8; ++r) {
      float mx = fmaxf(sc[0][r], sc[1][r]);
#pragma unroll
      for (int off = 8; off >= 1; off >>= 1)
        mx = fmaxf(mx, __shfl_xor(mx, off, 32));
      float mn = fmaxf(mrow[r], mx);
      float al = __expf(mrow[r] - mn);
      float p0 = __expf(sc[0][r] - mn);
      float p1 = __expf(sc[1][r] - mn);
      sc[0][r] = p0;
      sc[1][r] = p1;
      float rs = p0 + p1;
#pragma unroll
      for (int off = 8; off >= 1; off >>= 1)
        rs += __shfl_xor(rs, off, 32);
      lrow[r] = lrow[r] * al + rs;
      mrow[r] = mn;
#pragma unroll
      for (int dj = 0; dj < 4; ++dj) o[dj][r] *= al;
    }

    // ---- relayout P: C-layout -> A-layout via per-wave LDS bounce ----
#pragma unroll
    for (int j = 0; j < 2; ++j)
#pragma unroll
      for (int r = 0; r < 8; ++r)
        Ps[wave][(lh * 8 + r) * 32 + j * 16 + ml] = (_Float16)sc[j][r];

    asm volatile("s_wait_dscnt 0" ::: "memory");  // same-wave LDS RAW

    FragH pf;
    pf.u4[0] = *(const uint4*)&Ps[wave][ml * 32 + lh * 8];
    pf.u4[1] = *(const uint4*)&Ps[wave][ml * 32 + lh * 8 + 16];

    // ---- O += P @ V ----
#pragma unroll
    for (int dj = 0; dj < 4; ++dj) {
      FragH bv;
      int d = dj * 16 + ml;
      bv.u4[0] = *(const uint4*)&Vt[d * 32 + lh * 8];
      bv.u4[1] = *(const uint4*)&Vt[d * 32 + lh * 8 + 16];
      o[dj] = __builtin_amdgcn_wmma_f32_16x16x32_f16(
          false, pf.h, false, bv.h, (short)0, o[dj], false, false);
    }
  }

  // ---- normalize and store (merged-head layout) ----
  float linv[8];
#pragma unroll
  for (int r = 0; r < 8; ++r) linv[r] = 1.0f / lrow[r];
#pragma unroll
  for (int dj = 0; dj < 4; ++dj) {
    int col = dj * 16 + ml;
#pragma unroll
    for (int r = 0; r < 8; ++r) {
      int row = q0 + lh * 8 + r;
      O[(size_t)row * strideO + col] = o[dj][r] * linv[r];
    }
  }
}

// ---------------------------------------------------------------------------
// out = LayerNorm(x + a) * g + b    (D == 1024, one row per workgroup)
// ---------------------------------------------------------------------------
__global__ __launch_bounds__(256) void add_ln_kernel(
    const float* __restrict__ x, const float* __restrict__ a,
    const float* __restrict__ g, const float* __restrict__ bta,
    float* __restrict__ out, int D)
{
  const int row = blockIdx.x;
  const float* xr = x + (size_t)row * D;
  const float* ar = a + (size_t)row * D;
  float* orow = out + (size_t)row * D;

  float v[4];
  float s = 0.f, s2 = 0.f;
#pragma unroll
  for (int i = 0; i < 4; ++i) {
    int c = threadIdx.x + i * 256;
    float t = xr[c] + ar[c];
    v[i] = t; s += t; s2 += t * t;
  }
#pragma unroll
  for (int off = 16; off >= 1; off >>= 1) {
    s  += __shfl_xor(s,  off, 32);
    s2 += __shfl_xor(s2, off, 32);
  }
  __shared__ float sh[16];
  const int wv = threadIdx.x >> 5;
  if ((threadIdx.x & 31) == 0) { sh[wv] = s; sh[8 + wv] = s2; }
  __syncthreads();
  float ts = 0.f, ts2 = 0.f;
#pragma unroll
  for (int i = 0; i < 8; ++i) { ts += sh[i]; ts2 += sh[8 + i]; }
  const float invD = 1.0f / (float)D;
  const float mu   = ts * invD;
  const float var  = ts2 * invD - mu * mu;
  const float rstd = rsqrtf(var + 1e-5f);
#pragma unroll
  for (int i = 0; i < 4; ++i) {
    int c = threadIdx.x + i * 256;
    orow[c] = (v[i] - mu) * rstd * g[c] + bta[c];
  }
}

// ---------------------------------------------------------------------------
extern "C" void kernel_launch(void* const* d_in, const int* in_sizes, int n_in,
                              void* d_out, int out_size, void* d_ws, size_t ws_size,
                              hipStream_t stream)
{
  (void)in_sizes; (void)n_in; (void)out_size;
  const int B = 4, S = 1024, D = 1024, H = 16;
  const int BS = B * S;

  const float* x            = (const float*)d_in[0];
  const float* ctx          = (const float*)d_in[1];
  const float* c_attn_w     = (const float*)d_in[2];
  const float* c_attn_b     = (const float*)d_in[3];
  const float* self_proj_w  = (const float*)d_in[4];
  const float* self_proj_b  = (const float*)d_in[5];
  const float* q_w          = (const float*)d_in[6];
  const float* q_b          = (const float*)d_in[7];
  const float* kv_w         = (const float*)d_in[8];
  const float* kv_b         = (const float*)d_in[9];
  const float* cross_proj_w = (const float*)d_in[10];
  const float* cross_proj_b = (const float*)d_in[11];
  const float* fc_w         = (const float*)d_in[12];
  const float* fc_b         = (const float*)d_in[13];
  const float* mlp_proj_w   = (const float*)d_in[14];
  const float* mlp_proj_b   = (const float*)d_in[15];
  const float* ln1_g = (const float*)d_in[16];
  const float* ln1_b = (const float*)d_in[17];
  const float* ln2_g = (const float*)d_in[18];
  const float* ln2_b = (const float*)d_in[19];
  const float* ln3_g = (const float*)d_in[20];
  const float* ln3_b = (const float*)d_in[21];

  float* ws = (float*)d_ws;
  const size_t M1 = (size_t)1024 * 1024;
  if (ws_size < (size_t)40 * M1 * sizeof(float)) return;  // need 160 MB

  float* qkv  = ws;             // 12M floats [BS,3D] (later reused for q/kv)
  float* abuf = ws + 12 * M1;   // 4M  attention / mlp-proj output
  float* x1   = ws + 16 * M1;   // 4M  post-LN1
  float* x2   = ws + 20 * M1;   // 4M  post-LN2
  float* hbuf = ws + 24 * M1;   // 16M MLP hidden (head doubles as proj tmp)
  float* tmp  = hbuf;           // 4M  alias, consumed before hbuf is written
  float* qb   = qkv;            // 4M  cross-attn Q
  float* kvb  = qkv + 4 * M1;   // 8M  cross-attn KV

  dim3 blk256(256);
#define GEMM(Ain, Win, Bin, Cout, Mm, Nn, Kk, ACT) \
  gemm_f16_kernel<<<dim3((Nn) / 128, (Mm) / 128), blk256, 0, stream>>>( \
      Ain, Win, Bin, Cout, Mm, Nn, Kk, ACT)

  // --- causal self-attention ---
  GEMM(x, c_attn_w, c_attn_b, qkv, BS, 3 * D, D, 0);
  flash_attn_kernel<<<dim3(S / 64, B * H), dim3(128), 0, stream>>>(
      qkv, qkv + D, qkv + 2 * D, abuf, S, 3 * D, 3 * D, D, H, 1);
  GEMM(abuf, self_proj_w, self_proj_b, tmp, BS, D, D, 0);
  add_ln_kernel<<<dim3(BS), blk256, 0, stream>>>(x, tmp, ln1_g, ln1_b, x1, D);

  // --- cross-attention ---
  GEMM(x1, q_w, q_b, qb, BS, D, D, 0);
  GEMM(ctx, kv_w, kv_b, kvb, BS, 2 * D, D, 0);
  flash_attn_kernel<<<dim3(S / 64, B * H), dim3(128), 0, stream>>>(
      qb, kvb, kvb + D, abuf, S, D, 2 * D, D, H, 0);
  GEMM(abuf, cross_proj_w, cross_proj_b, tmp, BS, D, D, 0);
  add_ln_kernel<<<dim3(BS), blk256, 0, stream>>>(x1, tmp, ln2_g, ln2_b, x2, D);

  // --- MLP ---
  GEMM(x2, fc_w, fc_b, hbuf, BS, 4 * D, D, 1);            // fused GELU
  GEMM(hbuf, mlp_proj_w, mlp_proj_b, abuf, BS, D, 4 * D, 0);
  add_ln_kernel<<<dim3(BS), blk256, 0, stream>>>(x2, abuf, ln3_g, ln3_b,
                                                 (float*)d_out, D);
#undef GEMM
}
